// Graph_AutoEncoder_4518305595739
// MI455X (gfx1250) — matrix operand order repaired and checked by reading, not compiled, not run on hardware
//
#include <hip/hip_runtime.h>
#include <hip/hip_bf16.h>

// ---------------------------------------------------------------------------
// Graph AutoEncoder for MI455X (gfx1250), wave32 + WMMA f32<-f16.
// Persistent-workgroup RNN kernels: 16 batch rows per WGP, recurrent weights
// resident in the 320KB LDS as f16, v_wmma_f32_16x16x32_f16 for all GEMMs.
// All WMMA operands are kept in fragment-major (swizzled) layout so every
// fragment load is two 128-bit loads per lane (no scalar u16 gathers).
// ---------------------------------------------------------------------------

typedef __attribute__((ext_vector_type(16))) _Float16 v16h;
typedef __attribute__((ext_vector_type(8)))  float    v8f;

#define kB     2048
#define kT     200
#define MT     16      // batch rows per workgroup tile

__device__ __forceinline__ float sigf(float x) { return 1.0f / (1.0f + __expf(-x)); }

// ----- fragment-major layouts (ISA 7.12.2, wave32) --------------------------
// A fragment (16x32 of a 16xK tile): lane L holds row M=L%16;
//   half j -> k32 = ((L&16)?8:0) + (j&7) + ((j&8)?16:0)
// Swizzled A buffer: As[kt][lane][16 halfs]  (kt = k/32)
// Inverse map for scatter-stores of element (m, k):
__device__ __forceinline__ int a_sw_idx(int m, int k) {
  const int kt = k >> 5, k32 = k & 31;
  const int lane = m + ((k32 & 8) ? 16 : 0);
  const int j = (k32 & 7) | ((k32 & 16) ? 8 : 0);
  return (kt * 32 + lane) * 16 + j;
}
__device__ __forceinline__ v16h frag_a_ld(const _Float16* As, int kt, int lane) {
  return *(const v16h*)(As + (kt * 32 + lane) * 16);   // 32B contiguous -> 2x b128
}
// B fragment (32x16): lane L holds col N=L%16; half j -> k32 = ((L&16)?8:0) + (j&7) + ((j&8)?16:0)
// Swizzled B buffer: Bs[kt][nt][lane][16 halfs]
__device__ __forceinline__ v16h frag_b_ld(const _Float16* Bs, int ntiles, int kt, int nt, int lane) {
  return *(const v16h*)(Bs + (((kt * ntiles) + nt) * 32 + lane) * 16);
}
// C/D (16x16 f32): vgpr r, lanes 0-15 -> M=r, N=lane; lanes 16-31 -> M=8+r.
__device__ __forceinline__ void frag_c_store(float* C, int ldc, int n0, v8f acc, int lane) {
  const int n  = n0 + (lane & 15);
  const int mb = (lane & 16) ? 8 : 0;
#pragma unroll
  for (int r = 0; r < 8; ++r) C[(mb + r) * ldc + n] = acc[r];
}
__device__ __forceinline__ v8f v8f_zero() {
  v8f a;
#pragma unroll
  for (int i = 0; i < 8; ++i) a[i] = 0.0f;
  return a;
}
template <int KT>
__device__ __forceinline__ v8f gemm_sw(const _Float16* As, const _Float16* Bs,
                                       int ntiles, int nt, v8f acc, int lane) {
#pragma unroll
  for (int kt = 0; kt < KT; ++kt) {
    v16h a = frag_a_ld(As, kt, lane);
    v16h b = frag_b_ld(Bs, ntiles, kt, nt, lane);
    acc = __builtin_amdgcn_wmma_f32_16x16x32_f16(false, a, false, b, (short)0, acc, false, false);
  }
  return acc;
}

// ----- prep: W(f32, NOUT x K) -> fragment-major f16 Bs[kt][nt][lane][16] ----
__global__ void k_prep_b_sw(const float* __restrict__ W, _Float16* __restrict__ Bs,
                            int NOUT, int K, int ntiles, int ktiles) {
  int idx = blockIdx.x * blockDim.x + threadIdx.x;
  int total = ktiles * ntiles * 512;
  if (idx >= total) return;
  int j = idx & 15;
  int lane = (idx >> 4) & 31;
  int nt = (idx >> 9) % ntiles;
  int kt = (idx >> 9) / ntiles;
  int n = nt * 16 + (lane & 15);
  int k = kt * 32 + ((lane & 16) ? 8 : 0) + (j & 7) + ((j & 8) ? 16 : 0);
  float v = (n < NOUT && k < K) ? W[n * K + k] : 0.0f;
  Bs[idx] = (_Float16)v;
}

// ----- node embedding mean (B,2) -> (B,64) ----------------------------------
__global__ void k_embed_mean(const int* __restrict__ nd, const float* __restrict__ emb,
                             float* __restrict__ ne) {
  int b = blockIdx.x, e = threadIdx.x;
  int i0 = nd[b * 2 + 0], i1 = nd[b * 2 + 1];
  ne[(size_t)b * 64 + e] = 0.5f * (emb[(size_t)i0 * 64 + e] + emb[(size_t)i1 * 64 + e]);
}

// ----- BiLSTM layer 1: input dim 1, H=128; persistent over T ----------------
#define LDS1_BYTES (128*512*2 + 16*128*2 + 16*512*4 + 16*128*4)
__global__ void __launch_bounds__(128) k_lstm1(
    const float* __restrict__ edge,
    const float* __restrict__ wih_f, const float* __restrict__ bias_f,
    const float* __restrict__ wih_b, const float* __restrict__ bias_b,
    const _Float16* __restrict__ whhS_f, const _Float16* __restrict__ whhS_b,
    _Float16* __restrict__ h1sw, float* __restrict__ hn1) {
  extern __shared__ char smem[];
  _Float16* sW = (_Float16*)smem;            // swizzled whh^T: [4][32][32][16]
  _Float16* sH = sW + 128 * 512;             // swizzled A: [4][32][16]
  float* sGates = (float*)(sH + 16 * 128);   // [16][512]
  float* sC = sGates + 16 * 512;             // [16][128]

  const int tid = threadIdx.x, lane = tid & 31, wave = tid >> 5;
  const int tile = blockIdx.x, dir = blockIdx.y;
  const float* wih = dir ? wih_b : wih_f;
  const float* bias = dir ? bias_b : bias_f;
  const _Float16* whhS = dir ? whhS_b : whhS_f;

  { // bulk 16B copies of swizzled weights into LDS
    const uint4* s = (const uint4*)whhS;
    uint4* d = (uint4*)sW;
    for (int i = tid; i < 128 * 512 / 8; i += 128) d[i] = s[i];
  }
  for (int i = tid; i < 16 * 128; i += 128) { sH[i] = (_Float16)0.0f; sC[i] = 0.0f; }

  for (int ti = 0; ti < kT; ++ti) {
    const int t = dir ? (kT - 1 - ti) : ti;
    __syncthreads();
    for (int nt = wave * 8; nt < wave * 8 + 8; ++nt) {          // gates = h @ whh^T
      v8f acc = gemm_sw<4>(sH, sW, 32, nt, v8f_zero(), lane);
      frag_c_store(sGates, 512, nt * 16, acc, lane);
    }
    __syncthreads();
    for (int i = tid; i < 16 * 128; i += 128) {
      const int m = i >> 7, j = i & 127;
      const int gb = tile * MT + m;
      const float x = edge[(size_t)gb * kT + t];
      float gi = sGates[m * 512 + j]       + x * wih[j]       + bias[j];
      float gf = sGates[m * 512 + 128 + j] + x * wih[128 + j] + bias[128 + j];
      float gg = sGates[m * 512 + 256 + j] + x * wih[256 + j] + bias[256 + j];
      float go = sGates[m * 512 + 384 + j] + x * wih[384 + j] + bias[384 + j];
      const float ii = sigf(gi), ff = sigf(gf), g = tanhf(gg), oo = sigf(go);
      const float c = ff * sC[i] + ii * g;
      const float h = oo * tanhf(c);
      sC[i] = c;
      sH[a_sw_idx(m, j)] = (_Float16)h;
      // h1 output stored A-swizzled per (tile, t): 4096 halfs, K = dir*128+j
      h1sw[((size_t)tile * kT + t) * 4096 + a_sw_idx(m, dir * 128 + j)] = (_Float16)h;
      if (ti == kT - 1) hn1[(size_t)dir * kB * 128 + (size_t)gb * 128 + j] = h;
    }
  }
}

// ----- BiLSTM layer 2: input 256 (streamed swizzled f16), H=128 -------------
#define LDS2_BYTES (128*512*2 + 16*256*2 + 16*128*2 + 16*512*4 + 16*128*4)
__global__ void __launch_bounds__(128) k_lstm2(
    const _Float16* __restrict__ h1sw,
    const float* __restrict__ bias_f, const float* __restrict__ bias_b,
    const _Float16* __restrict__ wihS_f, const _Float16* __restrict__ wihS_b,
    const _Float16* __restrict__ whhS_f, const _Float16* __restrict__ whhS_b,
    float* __restrict__ hn2) {
  extern __shared__ char smem[];
  _Float16* sW = (_Float16*)smem;            // swizzled whh^T
  _Float16* sX = sW + 128 * 512;             // swizzled A: [8][32][16]
  _Float16* sH = sX + 16 * 256;              // swizzled A: [4][32][16]
  float* sGates = (float*)(sH + 16 * 128);   // [16][512]
  float* sC = sGates + 16 * 512;             // [16][128]

  const int tid = threadIdx.x, lane = tid & 31, wave = tid >> 5;
  const int tile = blockIdx.x, dir = blockIdx.y;
  const float* bias = dir ? bias_b : bias_f;
  const _Float16* wihS = dir ? wihS_b : wihS_f;   // [8][32][32][16] streamed from L2
  const _Float16* whhS = dir ? whhS_b : whhS_f;

  {
    const uint4* s = (const uint4*)whhS;
    uint4* d = (uint4*)sW;
    for (int i = tid; i < 128 * 512 / 8; i += 128) d[i] = s[i];
  }
  for (int i = tid; i < 16 * 128; i += 128) { sH[i] = (_Float16)0.0f; sC[i] = 0.0f; }

  for (int ti = 0; ti < kT; ++ti) {
    const int t = dir ? (kT - 1 - ti) : ti;
    __syncthreads();
    { // stage x_t: already fragment-major in h1sw -> contiguous 16B copies
      const uint4* s = (const uint4*)(h1sw + ((size_t)tile * kT + t) * 4096);
      uint4* d = (uint4*)sX;
      for (int i = tid; i < 4096 / 8; i += 128) d[i] = s[i];
    }
    __syncthreads();
    for (int nt = wave * 8; nt < wave * 8 + 8; ++nt) {  // x@wih^T + h@whh^T
      v8f acc = gemm_sw<8>(sX, wihS, 32, nt, v8f_zero(), lane);
      acc     = gemm_sw<4>(sH, sW,   32, nt, acc, lane);
      frag_c_store(sGates, 512, nt * 16, acc, lane);
    }
    __syncthreads();
    for (int i = tid; i < 16 * 128; i += 128) {
      const int m = i >> 7, j = i & 127;
      float gi = sGates[m * 512 + j]       + bias[j];
      float gf = sGates[m * 512 + 128 + j] + bias[128 + j];
      float gg = sGates[m * 512 + 256 + j] + bias[256 + j];
      float go = sGates[m * 512 + 384 + j] + bias[384 + j];
      const float ii = sigf(gi), ff = sigf(gf), g = tanhf(gg), oo = sigf(go);
      const float c = ff * sC[i] + ii * g;
      const float h = oo * tanhf(c);
      sC[i] = c;
      sH[a_sw_idx(m, j)] = (_Float16)h;
      if (ti == kT - 1)
        hn2[(size_t)dir * kB * 128 + ((size_t)tile * MT + m) * 128 + j] = h;
    }
  }
}

// ----- FC stack: (hn1+hn2 flat reshape) -> fc1 sigmoid -> fc2 sigmoid -> cat
#define LDSF_BYTES (16*256*2 + 16*256*2 + 16*256*4 + 16*64*4)
__global__ void __launch_bounds__(128) k_fc(
    const float* __restrict__ hn1, const float* __restrict__ hn2,
    const _Float16* __restrict__ fc1S, const _Float16* __restrict__ fc2S,
    const float* __restrict__ fc1_b, const float* __restrict__ fc2_b,
    const float* __restrict__ node_emb, float* __restrict__ hidcat) {
  extern __shared__ char smem[];
  _Float16* sA  = (_Float16*)smem;         // swizzled A: [8][32][16]
  _Float16* sA2 = sA + 16 * 256;           // swizzled A: [8][32][16]
  float* sO1 = (float*)(sA2 + 16 * 256);   // [16][256]
  float* sO2 = sO1 + 16 * 256;             // [16][64]

  const int tid = threadIdx.x, lane = tid & 31, wave = tid >> 5, tile = blockIdx.x;

  // reference does hn.reshape(-1,256) on a (2,B,128) stack -> pure flat indexing
  for (int i = tid; i < 16 * 256; i += 128) {
    const int m = i >> 8, c = i & 255;
    sA[a_sw_idx(m, c)] =
        (_Float16)(hn1[(size_t)tile * 16 * 256 + i] + hn2[(size_t)tile * 16 * 256 + i]);
  }
  __syncthreads();
  for (int nt = wave * 4; nt < wave * 4 + 4; ++nt) {
    v8f acc = gemm_sw<8>(sA, fc1S, 16, nt, v8f_zero(), lane);
    frag_c_store(sO1, 256, nt * 16, acc, lane);
  }
  __syncthreads();
  for (int i = tid; i < 16 * 256; i += 128) {
    const int m = i >> 8, c = i & 255;
    sA2[a_sw_idx(m, c)] = (_Float16)sigf(sO1[i] + fc1_b[c]);
  }
  __syncthreads();
  { // N = 64 -> 4 n-tiles, one per wave
    v8f acc = gemm_sw<8>(sA2, fc2S, 4, wave, v8f_zero(), lane);
    frag_c_store(sO2, 64, wave * 16, acc, lane);
  }
  __syncthreads();
  for (int i = tid; i < 16 * 128; i += 128) {
    const int m = i >> 7, c = i & 127;
    const size_t r = (size_t)tile * 16 + m;
    float v = (c < 64) ? sigf(sO2[m * 64 + c] + fc2_b[c]) : node_emb[r * 64 + (c - 64)];
    hidcat[r * 128 + c] = v;
  }
}

// ----- decoder: 200 steps of GRU(128) -> GRU(50 padded 64/160) -> sigmoid dec
#define LDSD_BYTES (128*384*2 + 128*160*2 + 64*160*2 + 16*128*2 + 16*64*2 \
                    + 16*128*4 + 16*64*4 + 16*384*4 + 16*160*4 + 16*160*4 + 16*4)
__global__ void __launch_bounds__(128) k_decoder(
    const float* __restrict__ edge, const float* __restrict__ hidcat,
    const _Float16* __restrict__ g1S, const _Float16* __restrict__ g2iS,
    const _Float16* __restrict__ g2hS,
    const float* __restrict__ g1_wih, const float* __restrict__ g1_bih,
    const float* __restrict__ g1_bhh,
    const float* __restrict__ g2_bih, const float* __restrict__ g2_bhh,
    const float* __restrict__ dec_w, const float* __restrict__ dec_b,
    float* __restrict__ out) {
  extern __shared__ char smem[];
  _Float16* sWg1  = (_Float16*)smem;       // swizzled [4][24][32][16]
  _Float16* sWg2i = sWg1 + 128 * 384;      // swizzled [4][10][32][16]
  _Float16* sWg2h = sWg2i + 128 * 160;     // swizzled [2][10][32][16]
  _Float16* sH1 = sWg2h + 64 * 160;        // swizzled A [4][32][16]
  _Float16* sH2 = sH1 + 16 * 128;          // swizzled A [2][32][16]
  float* sH1f = (float*)(sH2 + 16 * 64);   // [16][128]
  float* sH2f = sH1f + 16 * 128;           // [16][64]
  float* sG1  = sH2f + 16 * 64;            // [16][384]
  float* sGi2 = sG1 + 16 * 384;            // [16][160]
  float* sGh2 = sGi2 + 16 * 160;           // [16][160]
  float* sRes = sGh2 + 16 * 160;           // [16]

  const int tid = threadIdx.x, lane = tid & 31, wave = tid >> 5, tile = blockIdx.x;

  { const uint4* s = (const uint4*)g1S;  uint4* d = (uint4*)sWg1;
    for (int i = tid; i < 128 * 384 / 8; i += 128) d[i] = s[i]; }
  { const uint4* s = (const uint4*)g2iS; uint4* d = (uint4*)sWg2i;
    for (int i = tid; i < 128 * 160 / 8; i += 128) d[i] = s[i]; }
  { const uint4* s = (const uint4*)g2hS; uint4* d = (uint4*)sWg2h;
    for (int i = tid; i < 64 * 160 / 8; i += 128) d[i] = s[i]; }
  for (int i = tid; i < 16 * 128; i += 128) {
    const int m = i >> 7, j = i & 127;
    const float h = hidcat[(size_t)tile * 16 * 128 + i];
    sH1f[i] = h; sH1[a_sw_idx(m, j)] = (_Float16)h;
  }
  for (int i = tid; i < 16 * 64; i += 128) { sH2f[i] = 0.0f; sH2[i] = (_Float16)0.0f; }
  if (tid < 16) sRes[tid] = edge[(size_t)(tile * MT + tid) * kT + (kT - 1)];

  for (int t = 0; t < kT; ++t) {
    __syncthreads();
    for (int nt = wave; nt < 24; nt += 4) {           // gh1 = h1 @ g1_whh^T
      v8f acc = gemm_sw<4>(sH1, sWg1, 24, nt, v8f_zero(), lane);
      frag_c_store(sG1, 384, nt * 16, acc, lane);
    }
    __syncthreads();
    for (int i = tid; i < 16 * 128; i += 128) {       // GRU1 (input dim 1)
      const int m = i >> 7, j = i & 127;
      const float res = sRes[m];
      const float ir = res * g1_wih[j]       + g1_bih[j];
      const float iz = res * g1_wih[128 + j] + g1_bih[128 + j];
      const float in_ = res * g1_wih[256 + j] + g1_bih[256 + j];
      const float hr = sG1[m * 384 + j]       + g1_bhh[j];
      const float hz = sG1[m * 384 + 128 + j] + g1_bhh[128 + j];
      const float hn = sG1[m * 384 + 256 + j] + g1_bhh[256 + j];
      const float r = sigf(ir + hr), z = sigf(iz + hz);
      const float n = tanhf(in_ + r * hn);
      const float h = (1.0f - z) * n + z * sH1f[i];
      sH1f[i] = h; sH1[a_sw_idx(m, j)] = (_Float16)h;
    }
    __syncthreads();
    for (int nt = wave; nt < 10; nt += 4) {           // gi2, gh2 (N pad 160)
      v8f a1 = gemm_sw<4>(sH1, sWg2i, 10, nt, v8f_zero(), lane);
      frag_c_store(sGi2, 160, nt * 16, a1, lane);
      v8f a2 = gemm_sw<2>(sH2, sWg2h, 10, nt, v8f_zero(), lane);
      frag_c_store(sGh2, 160, nt * 16, a2, lane);
    }
    __syncthreads();
    for (int i = tid; i < 16 * 50; i += 128) {        // GRU2 (50 valid cols)
      const int m = i / 50, j = i % 50;
      const float ir = sGi2[m * 160 + j]        + g2_bih[j];
      const float iz = sGi2[m * 160 + 50 + j]   + g2_bih[50 + j];
      const float in_ = sGi2[m * 160 + 100 + j] + g2_bih[100 + j];
      const float hr = sGh2[m * 160 + j]        + g2_bhh[j];
      const float hz = sGh2[m * 160 + 50 + j]   + g2_bhh[50 + j];
      const float hn = sGh2[m * 160 + 100 + j]  + g2_bhh[100 + j];
      const float r = sigf(ir + hr), z = sigf(iz + hz);
      const float n = tanhf(in_ + r * hn);
      const float h = (1.0f - z) * n + z * sH2f[m * 64 + j];
      sH2f[m * 64 + j] = h; sH2[a_sw_idx(m, j)] = (_Float16)h;
    }
    __syncthreads();
    if (tid < 16) {                                   // dec: 50-dot + sigmoid
      float acc = dec_b[0];
      for (int j = 0; j < 50; ++j) acc += sH2f[tid * 64 + j] * dec_w[j];
      const float rr = sigf(acc);
      sRes[tid] = rr;
      out[(size_t)(tile * MT + tid) * kT + t] = rr;
    }
  }
}

// ---------------------------------------------------------------------------
extern "C" void kernel_launch(void* const* d_in, const int* in_sizes, int n_in,
                              void* d_out, int out_size, void* d_ws, size_t ws_size,
                              hipStream_t stream) {
  (void)in_sizes; (void)n_in; (void)out_size; (void)ws_size;
  const int*   node_data = (const int*)  d_in[0];
  const float* edge      = (const float*)d_in[1];
  const float* emb       = (const float*)d_in[2];
  const float* l1_wih_f  = (const float*)d_in[3];
  const float* l1_whh_f  = (const float*)d_in[4];
  const float* l1_b_f    = (const float*)d_in[5];
  const float* l1_wih_b  = (const float*)d_in[6];
  const float* l1_whh_b  = (const float*)d_in[7];
  const float* l1_b_b    = (const float*)d_in[8];
  const float* l2_wih_f  = (const float*)d_in[9];
  const float* l2_whh_f  = (const float*)d_in[10];
  const float* l2_b_f    = (const float*)d_in[11];
  const float* l2_wih_b  = (const float*)d_in[12];
  const float* l2_whh_b  = (const float*)d_in[13];
  const float* l2_b_b    = (const float*)d_in[14];
  const float* fc1_w     = (const float*)d_in[15];
  const float* fc1_b     = (const float*)d_in[16];
  const float* fc2_w     = (const float*)d_in[17];
  const float* fc2_b     = (const float*)d_in[18];
  const float* g1_wih    = (const float*)d_in[19];
  const float* g1_whh    = (const float*)d_in[20];
  const float* g1_bih    = (const float*)d_in[21];
  const float* g1_bhh    = (const float*)d_in[22];
  const float* g2_wih    = (const float*)d_in[23];
  const float* g2_whh    = (const float*)d_in[24];
  const float* g2_bih    = (const float*)d_in[25];
  const float* g2_bhh    = (const float*)d_in[26];
  const float* dec_w     = (const float*)d_in[27];
  const float* dec_b     = (const float*)d_in[28];

  char* ws = (char*)d_ws;
  size_t off = 0;
  auto take = [&](size_t bytes) -> void* {
    void* p = ws + off;
    off += (bytes + 255) & ~(size_t)255;
    return p;
  };
  float* w_node_emb = (float*)take((size_t)kB * 64 * 4);
  float* w_hn1      = (float*)take((size_t)2 * kB * 128 * 4);
  float* w_hn2      = (float*)take((size_t)2 * kB * 128 * 4);
  float* w_hidcat   = (float*)take((size_t)kB * 128 * 4);
  _Float16* w_l1whhS_f = (_Float16*)take(128 * 512 * 2);
  _Float16* w_l1whhS_b = (_Float16*)take(128 * 512 * 2);
  _Float16* w_l2wihS_f = (_Float16*)take(256 * 512 * 2);
  _Float16* w_l2wihS_b = (_Float16*)take(256 * 512 * 2);
  _Float16* w_l2whhS_f = (_Float16*)take(128 * 512 * 2);
  _Float16* w_l2whhS_b = (_Float16*)take(128 * 512 * 2);
  _Float16* w_fc1S     = (_Float16*)take(256 * 256 * 2);
  _Float16* w_fc2S     = (_Float16*)take(256 * 64 * 2);
  _Float16* w_g1S      = (_Float16*)take(128 * 384 * 2);
  _Float16* w_g2iS     = (_Float16*)take(128 * 160 * 2);
  _Float16* w_g2hS     = (_Float16*)take(64 * 160 * 2);
  _Float16* w_h1sw     = (_Float16*)take((size_t)kB * kT * 256 * 2);  // 210MB f16

  auto prep = [&](const float* W, _Float16* Bs, int NOUT, int K, int ntiles, int ktiles) {
    int n = ktiles * ntiles * 512;
    k_prep_b_sw<<<(n + 255) / 256, 256, 0, stream>>>(W, Bs, NOUT, K, ntiles, ktiles);
  };
  prep(l1_whh_f, w_l1whhS_f, 512, 128, 32, 4);
  prep(l1_whh_b, w_l1whhS_b, 512, 128, 32, 4);
  prep(l2_wih_f, w_l2wihS_f, 512, 256, 32, 8);
  prep(l2_wih_b, w_l2wihS_b, 512, 256, 32, 8);
  prep(l2_whh_f, w_l2whhS_f, 512, 128, 32, 4);
  prep(l2_whh_b, w_l2whhS_b, 512, 128, 32, 4);
  prep(fc1_w,    w_fc1S,     256, 256, 16, 8);
  prep(fc2_w,    w_fc2S,      64, 256,  4, 8);
  prep(g1_whh,   w_g1S,      384, 128, 24, 4);
  prep(g2_wih,   w_g2iS,     150, 128, 10, 4);
  prep(g2_whh,   w_g2hS,     150,  50, 10, 2);

  k_embed_mean<<<kB, 64, 0, stream>>>(node_data, emb, w_node_emb);

  hipFuncSetAttribute((const void*)k_lstm1, hipFuncAttributeMaxDynamicSharedMemorySize, LDS1_BYTES);
  k_lstm1<<<dim3(kB / MT, 2), 128, LDS1_BYTES, stream>>>(
      edge, l1_wih_f, l1_b_f, l1_wih_b, l1_b_b,
      w_l1whhS_f, w_l1whhS_b, w_h1sw, w_hn1);

  hipFuncSetAttribute((const void*)k_lstm2, hipFuncAttributeMaxDynamicSharedMemorySize, LDS2_BYTES);
  k_lstm2<<<dim3(kB / MT, 2), 128, LDS2_BYTES, stream>>>(
      w_h1sw, l2_b_f, l2_b_b, w_l2wihS_f, w_l2wihS_b, w_l2whhS_f, w_l2whhS_b, w_hn2);

  k_fc<<<kB / MT, 128, LDSF_BYTES, stream>>>(
      w_hn1, w_hn2, w_fc1S, w_fc2S, fc1_b, fc2_b, w_node_emb, w_hidcat);

  hipFuncSetAttribute((const void*)k_decoder, hipFuncAttributeMaxDynamicSharedMemorySize, LDSD_BYTES);
  k_decoder<<<kB / MT, 128, LDSD_BYTES, stream>>>(
      edge, w_hidcat, w_g1S, w_g2iS, w_g2hS,
      g1_wih, g1_bih, g1_bhh, g2_bih, g2_bhh, dec_w, dec_b, (float*)d_out);
}
